// MyBlock_62182536512193
// MI455X (gfx1250) — matrix operand order repaired and checked by reading, not compiled, not run on hardware
//
#include <hip/hip_runtime.h>
#include <cstdint>

typedef __attribute__((ext_vector_type(16))) _Float16 v16h;
typedef __attribute__((ext_vector_type(8)))  float    v8f;

static constexpr int  Bn  = 8;
static constexpr int  Ntk = 4096;    // tokens N
static constexpr int  Np0 = 16384;   // N0 points
static constexpr int  Cch = 64;      // C
static constexpr int  HID = 256;
static constexpr int  Hh  = 128, Ww = 128, HW = Hh * Ww;
static constexpr int  R2  = 256;     // (H/SR)*(W/SR)
static constexpr float EPS = 1e-6f;

// ---------------------------------------------------------------- elementwise

__global__ void k_zero(float* __restrict__ p, long n) {
  long i = (long)blockIdx.x * blockDim.x + threadIdx.x;
  long st = (long)gridDim.x * blockDim.x;
  for (; i < n; i += st) p[i] = 0.0f;
}

__global__ void k_cvt(_Float16* __restrict__ dst, const float* __restrict__ src, long n) {
  long i = (long)blockIdx.x * blockDim.x + threadIdx.x;
  if (i < n) dst[i] = (_Float16)src[i];
}

// src (Nr,Kc) row-major -> dst (Kc,Nr):  dst[k*Nr+n] = src[n*Kc+k]
__global__ void k_tcvt(_Float16* __restrict__ dst, const float* __restrict__ src, int Nr, int Kc) {
  long i = (long)blockIdx.x * blockDim.x + threadIdx.x;
  if (i >= (long)Nr * Kc) return;
  int n = (int)(i / Kc), k = (int)(i % Kc);
  dst[(long)k * Nr + n] = (_Float16)src[i];
}

// pixel_index + per-pixel counts
__global__ void k_pix(const float* __restrict__ loc, int* __restrict__ pix, float* __restrict__ cnt) {
  int i = blockIdx.x * blockDim.x + threadIdx.x;
  if (i >= Bn * Np0) return;
  float lx = loc[2 * i], ly = loc[2 * i + 1];
  lx = fminf(fmaxf(lx, -1.f), 1.f);
  ly = fminf(fmaxf(ly, -1.f), 1.f);
  lx = (lx + 1.f) * 0.5f;
  ly = (ly + 1.f) * 0.5f;
  int xi = (int)roundf(lx * (Ww - 1)); xi = xi < 0 ? 0 : (xi > Ww - 1 ? Ww - 1 : xi);
  int yi = (int)roundf(ly * (Hh - 1)); yi = yi < 0 ? 0 : (yi > Hh - 1 ? Hh - 1 : yi);
  int p = yi * Ww + xi;
  pix[i] = p;
  int b = i / Np0;
  atomicAdd(&cnt[(long)b * HW + p], 1.0f);
}

// layernorm over 64 channels, one wave per token, f16 out
__global__ void k_ln64(const float* __restrict__ src, const float* __restrict__ gw,
                       const float* __restrict__ gb, _Float16* __restrict__ dst, int ntok) {
  int wid = blockIdx.x * (blockDim.x >> 5) + (threadIdx.x >> 5);
  int lane = threadIdx.x & 31;
  if (wid >= ntok) return;
  const float* r = src + (long)wid * 64;
  float v0 = r[lane], v1 = r[lane + 32];
  float s = v0 + v1;
  for (int o = 16; o > 0; o >>= 1) s += __shfl_xor(s, o);
  float m = s * (1.f / 64.f);
  float d0 = v0 - m, d1 = v1 - m;
  float q = d0 * d0 + d1 * d1;
  for (int o = 16; o > 0; o >>= 1) q += __shfl_xor(q, o);
  float inv = rsqrtf(q * (1.f / 64.f) + 1e-5f);
  dst[(long)wid * 64 + lane]      = (_Float16)(d0 * inv * gw[lane] + gb[lane]);
  dst[(long)wid * 64 + lane + 32] = (_Float16)(d1 * inv * gw[lane + 32] + gb[lane + 32]);
}

// --------------------------------------------------------------- WMMA GEMM
// One wave computes one 16x16 tile of C = alpha*A(f16,MxK)@B(f16,KxN) + bias + resid.
// Fragment layouts per CDNA5 ISA 7.12.2 (16-bit A 16x32; B 32x16; f32 C/D).
__global__ void k_wmma_gemm(const _Float16* __restrict__ A, int lda, long sA,
                            const _Float16* __restrict__ Bm, int ldb, long sB,
                            const float* __restrict__ bias,
                            const float* __restrict__ resid, long sR,
                            float* __restrict__ outF,
                            _Float16* __restrict__ outH,
                            int ldc, long sC, int K, float alpha) {
  int m0 = blockIdx.x * 16, n0 = blockIdx.y * 16, b = blockIdx.z;
  const _Float16* Ab = A + (long)b * sA;
  const _Float16* Bb = Bm + (long)b * sB;
  int l = threadIdx.x, lo = l & 15, hi = l >> 4;
  v8f acc = {0.f, 0.f, 0.f, 0.f, 0.f, 0.f, 0.f, 0.f};
  const _Float16* ap = Ab + (long)(m0 + lo) * lda + (hi ? 8 : 0);
  for (int kb = 0; kb < K; kb += 32) {
    union { v16h v; uint32_t u[8]; } af, bf;
#pragma unroll
    for (int j = 0; j < 8; ++j) {
      int koff = (j < 4) ? (2 * j) : (16 + 2 * (j - 4));
      af.u[j] = *(const uint32_t*)(ap + kb + koff);
    }
    const _Float16* bp = Bb + (long)(kb + (hi ? 8 : 0)) * ldb + n0 + lo;
#pragma unroll
    for (int s = 0; s < 16; ++s) bf.v[s] = bp[(long)((s < 8) ? s : (s + 8)) * ldb];
    acc = __builtin_amdgcn_wmma_f32_16x16x32_f16(false, af.v, false, bf.v,
                                                 (short)0, acc, false, false);
  }
  int n = n0 + lo;
  float bv = bias ? bias[n] : 0.f;
#pragma unroll
  for (int r = 0; r < 8; ++r) {
    int m = m0 + r + (hi ? 8 : 0);
    long idx = (long)b * sC + (long)m * ldc + n;
    float v = acc[r] * alpha + bv;
    if (resid) v += resid[(long)b * sR + (long)m * ldc + n];
    if (outF) outF[idx] = v;
    if (outH) outH[idx] = (_Float16)v;
  }
}

// ----------------------------------------------------------- scatter/gather

// token2map scatter, 64-channel f16 source (xn)
__global__ void k_scatter64(const _Float16* __restrict__ tok, const int* __restrict__ idx,
                            const int* __restrict__ pix, float* __restrict__ map) {
  long g = (long)blockIdx.x * blockDim.x + threadIdx.x;
  if (g >= (long)Bn * Np0 * 64) return;
  int c = (int)(g & 63);
  long pi = g >> 6;
  int b = (int)(pi / Np0);
  int t = idx[pi];
  int p = pix[pi];
  atomicAdd(&map[((long)b * HW + p) * 64 + c], (float)tok[((long)b * Ntk + t) * 64 + c]);
}

// token2map scatter, 256-channel f32 source (h)
__global__ void k_scatter256(const float* __restrict__ tok, const int* __restrict__ idx,
                             const int* __restrict__ pix, float* __restrict__ map) {
  long g = (long)blockIdx.x * blockDim.x + threadIdx.x;
  if (g >= (long)Bn * Np0 * 256) return;
  int c = (int)(g & 255);
  long pi = g >> 8;
  int b = (int)(pi / Np0);
  int t = idx[pi];
  int p = pix[pi];
  atomicAdd(&map[((long)b * HW + p) * 256 + c], tok[((long)b * Ntk + t) * 256 + c]);
}

// extract stride-8 non-overlapping 8x8 patches (normalized by counts) -> f16 GEMM A
__global__ void k_patches(const float* __restrict__ xmap, const float* __restrict__ cnt,
                          _Float16* __restrict__ patches) {
  long g = (long)blockIdx.x * blockDim.x + threadIdx.x;
  if (g >= (long)Bn * R2 * 4096) return;
  int e = (int)(g & 4095);
  long t = g >> 12;
  int op = (int)(t & 255);
  int b = (int)(t >> 8);
  int c = e & 63, dx = (e >> 6) & 7, dy = e >> 9;
  int ox = op & 15, oy = op >> 4;
  int p = (oy * 8 + dy) * Ww + ox * 8 + dx;
  float v = xmap[((long)b * HW + p) * 64 + c] / (cnt[(long)b * HW + p] + EPS);
  patches[g] = (_Float16)v;
}

// split kv (B,256,128) -> kT (B,64,256) and v (B,256,64)
__global__ void k_repack_kv(const _Float16* __restrict__ kv, _Float16* __restrict__ kT,
                            _Float16* __restrict__ vv) {
  long g = (long)blockIdx.x * blockDim.x + threadIdx.x;
  if (g >= (long)Bn * R2 * 128) return;
  int c = (int)(g & 127);
  long t = g >> 7;
  int r = (int)(t & 255);
  int b = (int)(t >> 8);
  _Float16 val = kv[g];
  if (c < 64) kT[((long)b * 64 + c) * R2 + r] = val;
  else        vv[((long)b * R2 + r) * 64 + (c - 64)] = val;
}

// softmax over 256, one wave per row, f16 probs out
__global__ void k_softmax(const float* __restrict__ s, _Float16* __restrict__ pr, int rows) {
  int wid = blockIdx.x * (blockDim.x >> 5) + (threadIdx.x >> 5);
  int lane = threadIdx.x & 31;
  if (wid >= rows) return;
  const float* r = s + (long)wid * 256;
  float v[8];
  float mx = -1e30f;
#pragma unroll
  for (int j = 0; j < 8; ++j) { v[j] = r[j * 32 + lane]; mx = fmaxf(mx, v[j]); }
  for (int o = 16; o > 0; o >>= 1) mx = fmaxf(mx, __shfl_xor(mx, o));
  float sum = 0.f;
#pragma unroll
  for (int j = 0; j < 8; ++j) { v[j] = __expf(v[j] - mx); sum += v[j]; }
  for (int o = 16; o > 0; o >>= 1) sum += __shfl_xor(sum, o);
  float inv = 1.f / sum;
  _Float16* o_ = pr + (long)wid * 256;
#pragma unroll
  for (int j = 0; j < 8; ++j) o_[j * 32 + lane] = (_Float16)(v[j] * inv);
}

// normalize hmap by counts
__global__ void k_normmap(float* __restrict__ hmap, const float* __restrict__ cnt) {
  long g = (long)blockIdx.x * blockDim.x + threadIdx.x;
  if (g >= (long)Bn * HW * 256) return;
  hmap[g] /= (cnt[g >> 8] + EPS);
}

// scatter agg weights per token
__global__ void k_sw(const float* __restrict__ w, const int* __restrict__ idx,
                     float* __restrict__ sw) {
  int i = blockIdx.x * blockDim.x + threadIdx.x;
  if (i >= Bn * Np0) return;
  int b = i / Np0;
  atomicAdd(&sw[(long)b * Ntk + idx[i]], w[i]);
}

// fused depthwise 3x3 conv + map2token_nearest accumulate
__global__ void k_dwm2t(const float* __restrict__ hmap, const float* __restrict__ dww,
                        const float* __restrict__ dwb, const int* __restrict__ pix,
                        const int* __restrict__ idx, const float* __restrict__ w,
                        const float* __restrict__ sw, float* __restrict__ mt) {
  long g = (long)blockIdx.x * blockDim.x + threadIdx.x;
  if (g >= (long)Bn * Np0 * 256) return;
  int c = (int)(g & 255);
  long pi = g >> 8;
  int b = (int)(pi / Np0);
  int p = pix[pi];
  int y = p >> 7, x = p & 127;
  float acc = dwb[c];
#pragma unroll
  for (int dy = 0; dy < 3; ++dy) {
    int yy = y + dy - 1;
    if ((unsigned)yy >= (unsigned)Hh) continue;
#pragma unroll
    for (int dx = 0; dx < 3; ++dx) {
      int xx = x + dx - 1;
      if ((unsigned)xx >= (unsigned)Ww) continue;
      acc += hmap[((long)b * HW + yy * Ww + xx) * 256 + c] * dww[(dy * 3 + dx) * 256 + c];
    }
  }
  int t = idx[pi];
  float nw = w[pi] / (sw[(long)b * Ntk + t] + EPS);
  atomicAdd(&mt[((long)b * Ntk + t) * 256 + c], acc * nw);
}

// h2 = gelu(mt + h*dwskip) -> f16
__global__ void k_h2(const float* __restrict__ mt, const float* __restrict__ h,
                     const float* __restrict__ skip, _Float16* __restrict__ out) {
  long g = (long)blockIdx.x * blockDim.x + threadIdx.x;
  if (g >= (long)Bn * Ntk * 256) return;
  int c = (int)(g & 255);
  float v = mt[g] + h[g] * skip[c];
  out[g] = (_Float16)(0.5f * v * (1.0f + erff(v * 0.70710678118f)));
}

// ------------------------------------------------------------------- driver

extern "C" void kernel_launch(void* const* d_in, const int* in_sizes, int n_in,
                              void* d_out, int out_size, void* d_ws, size_t ws_size,
                              hipStream_t stream) {
  (void)in_sizes; (void)n_in; (void)out_size; (void)ws_size;
  const float* x      = (const float*)d_in[0];
  const float* loc    = (const float*)d_in[1];
  const int*   idxagg = (const int*)d_in[2];
  const float* aggw   = (const float*)d_in[3];
  const float* n1w = (const float*)d_in[6];
  const float* n1b = (const float*)d_in[7];
  const float* n2w = (const float*)d_in[8];
  const float* n2b = (const float*)d_in[9];
  const float* qw  = (const float*)d_in[10];
  const float* qb  = (const float*)d_in[11];
  const float* kvw = (const float*)d_in[12];
  const float* kvb = (const float*)d_in[13];
  const float* pw  = (const float*)d_in[14];
  const float* pb  = (const float*)d_in[15];
  const float* srw = (const float*)d_in[16];
  const float* srb = (const float*)d_in[17];
  const float* srnw = (const float*)d_in[18];
  const float* srnb = (const float*)d_in[19];
  const float* f1w = (const float*)d_in[20];
  const float* f1b = (const float*)d_in[21];
  const float* dww = (const float*)d_in[22];
  const float* dwb = (const float*)d_in[23];
  const float* dwsk = (const float*)d_in[24];
  const float* f2w = (const float*)d_in[25];
  const float* f2b = (const float*)d_in[26];
  float* out = (float*)d_out;

  char* wp = (char*)d_ws;
  auto alloc = [&](size_t bytes) -> char* {
    char* r = wp;
    wp += (bytes + 255) & ~(size_t)255;
    return r;
  };

  int*      pix   = (int*)alloc((size_t)Bn * Np0 * 4);
  float*    cnt   = (float*)alloc((size_t)Bn * HW * 4);
  float*    sw    = (float*)alloc((size_t)Bn * Ntk * 4);
  _Float16* xn_h  = (_Float16*)alloc((size_t)Bn * Ntk * 64 * 2);
  _Float16* q_h   = (_Float16*)alloc((size_t)Bn * Ntk * 64 * 2);
  _Float16* kv_h  = (_Float16*)alloc((size_t)Bn * R2 * 128 * 2);
  _Float16* kT_h  = (_Float16*)alloc((size_t)Bn * 64 * R2 * 2);
  _Float16* v_h   = (_Float16*)alloc((size_t)Bn * R2 * 64 * 2);
  float*    xr    = (float*)alloc((size_t)Bn * R2 * 64 * 4);
  _Float16* xrn_h = (_Float16*)alloc((size_t)Bn * R2 * 64 * 2);
  _Float16* ao_h  = (_Float16*)alloc((size_t)Bn * Ntk * 64 * 2);
  float*    x1    = (float*)alloc((size_t)Bn * Ntk * 64 * 4);
  _Float16* x1n_h = (_Float16*)alloc((size_t)Bn * Ntk * 64 * 2);
  float*    hbuf  = (float*)alloc((size_t)Bn * Ntk * 256 * 4);
  float*    mt    = (float*)alloc((size_t)Bn * Ntk * 256 * 4);
  _Float16* h2_h  = (_Float16*)alloc((size_t)Bn * Ntk * 256 * 2);
  _Float16* qw_h  = (_Float16*)alloc(64 * 64 * 2);
  _Float16* kvw_h = (_Float16*)alloc(64 * 128 * 2);
  _Float16* pw_h  = (_Float16*)alloc(64 * 64 * 2);
  _Float16* srw_h = (_Float16*)alloc(4096 * 64 * 2);
  _Float16* f1w_h = (_Float16*)alloc(64 * 256 * 2);
  _Float16* f2w_h = (_Float16*)alloc(256 * 64 * 2);
  // 134MB arena shared by: (xmap, patches) -> (scores, probs) -> hmap
  char* arena = alloc((size_t)Bn * HW * 256 * 4);
  float*    hmap    = (float*)arena;
  float*    xmap    = (float*)arena;
  _Float16* patches = (_Float16*)(arena + (size_t)Bn * HW * 64 * 4);
  float*    scores  = (float*)arena;
  _Float16* probs   = (_Float16*)(arena + (size_t)Bn * Ntk * 256 * 4);

  auto blks = [](long n) { return (unsigned)((n + 255) / 256); };
  auto gemm = [&](const _Float16* A, int lda, long sA, const _Float16* Bm, int ldb, long sB,
                  const float* bias, const float* resid, long sR,
                  float* oF, _Float16* oH, int ldc, long sC, int M, int N, int K, float alpha) {
    dim3 grid(M / 16, N / 16, Bn);
    k_wmma_gemm<<<grid, 32, 0, stream>>>(A, lda, sA, Bm, ldb, sB, bias, resid, sR,
                                         oF, oH, ldc, sC, K, alpha);
  };

  // init accumulators
  k_zero<<<2048, 256, 0, stream>>>(cnt, (long)Bn * HW);
  k_zero<<<2048, 256, 0, stream>>>(sw, (long)Bn * Ntk);
  k_zero<<<2048, 256, 0, stream>>>(xmap, (long)Bn * HW * 64);
  k_zero<<<2048, 256, 0, stream>>>(mt, (long)Bn * Ntk * 256);

  // f16 weight repacks: B matrices stored (K,N) row-major
  k_tcvt<<<blks(64 * 64), 256, 0, stream>>>(qw_h, qw, 64, 64);
  k_tcvt<<<blks(128 * 64), 256, 0, stream>>>(kvw_h, kvw, 128, 64);
  k_tcvt<<<blks(64 * 64), 256, 0, stream>>>(pw_h, pw, 64, 64);
  k_cvt<<<blks(4096 * 64), 256, 0, stream>>>(srw_h, srw, 4096L * 64);  // HWIO already (K,N)
  k_tcvt<<<blks(256 * 64), 256, 0, stream>>>(f1w_h, f1w, 256, 64);
  k_tcvt<<<blks(64 * 256), 256, 0, stream>>>(f2w_h, f2w, 64, 256);

  // pix indices + counts; agg-weight sums
  k_pix<<<blks((long)Bn * Np0), 256, 0, stream>>>(loc, pix, cnt);
  k_sw<<<blks((long)Bn * Np0), 256, 0, stream>>>(aggw, idxagg, sw);

  // xn = LN(x); q = xn@qw^T+qb
  k_ln64<<<blks((long)Bn * Ntk * 32), 256, 0, stream>>>(x, n1w, n1b, xn_h, Bn * Ntk);
  gemm(xn_h, 64, (long)Ntk * 64, qw_h, 64, 0, qb, nullptr, 0,
       nullptr, q_h, 64, (long)Ntk * 64, Ntk, 64, 64, 1.f);

  // token2map(xn) -> xmap ; patch extraction (÷cnt) ; sr conv as GEMM
  k_scatter64<<<blks((long)Bn * Np0 * 64), 256, 0, stream>>>(xn_h, idxagg, pix, xmap);
  k_patches<<<blks((long)Bn * R2 * 4096), 256, 0, stream>>>(xmap, cnt, patches);
  gemm(patches, 4096, (long)R2 * 4096, srw_h, 64, 0, srb, nullptr, 0,
       xr, nullptr, 64, (long)R2 * 64, R2, 64, 4096, 1.f);

  // xr LN ; kv GEMM ; split k^T / v
  k_ln64<<<blks((long)Bn * R2 * 32), 256, 0, stream>>>(xr, srnw, srnb, xrn_h, Bn * R2);
  gemm(xrn_h, 64, (long)R2 * 64, kvw_h, 128, 0, kvb, nullptr, 0,
       nullptr, kv_h, 128, (long)R2 * 128, R2, 128, 64, 1.f);
  k_repack_kv<<<blks((long)Bn * R2 * 128), 256, 0, stream>>>(kv_h, kT_h, v_h);

  // attention (conf_r == 0 in reference, so no bias)
  gemm(q_h, 64, (long)Ntk * 64, kT_h, R2, (long)64 * R2, nullptr, nullptr, 0,
       scores, nullptr, R2, (long)Ntk * R2, Ntk, R2, 64, 0.125f);
  k_softmax<<<blks((long)Bn * Ntk * 32), 256, 0, stream>>>(scores, probs, Bn * Ntk);
  gemm(probs, R2, (long)Ntk * R2, v_h, 64, (long)R2 * 64, nullptr, nullptr, 0,
       nullptr, ao_h, 64, (long)Ntk * 64, Ntk, 64, R2, 1.f);

  // x1 = x + proj(ao) ; h = LN(x1)@fc1^T + b
  gemm(ao_h, 64, (long)Ntk * 64, pw_h, 64, 0, pb, x, (long)Ntk * 64,
       x1, nullptr, 64, (long)Ntk * 64, Ntk, 64, 64, 1.f);
  k_ln64<<<blks((long)Bn * Ntk * 32), 256, 0, stream>>>(x1, n2w, n2b, x1n_h, Bn * Ntk);
  gemm(x1n_h, 64, (long)Ntk * 64, f1w_h, 256, 0, f1b, nullptr, 0,
       hbuf, nullptr, 256, (long)Ntk * 256, Ntk, 256, 64, 1.f);

  // token2map(h) -> hmap (arena now free of scores/probs), normalize,
  // fused depthwise-3x3 + map2token, skip + gelu
  k_zero<<<8192, 256, 0, stream>>>(hmap, (long)Bn * HW * 256);
  k_scatter256<<<blks((long)Bn * Np0 * 256), 256, 0, stream>>>(hbuf, idxagg, pix, hmap);
  k_normmap<<<blks((long)Bn * HW * 256), 256, 0, stream>>>(hmap, cnt);
  k_dwm2t<<<blks((long)Bn * Np0 * 256), 256, 0, stream>>>(hmap, dww, dwb, pix, idxagg,
                                                          aggw, sw, mt);
  k_h2<<<blks((long)Bn * Ntk * 256), 256, 0, stream>>>(mt, hbuf, dwsk, h2_h);

  // x2 = x1 + fc2(h2)
  gemm(h2_h, 256, (long)Ntk * 256, f2w_h, 64, 0, f2b, x1, (long)Ntk * 64,
       out, nullptr, 64, (long)Ntk * 64, Ntk, 64, 256, 1.f);
}
